// GNSSolver_32701880992368
// MI455X (gfx1250) — matrix operand order repaired and checked by reading, not compiled, not run on hardware
//
#include <hip/hip_runtime.h>
#include <cstdint>
#include <cstddef>

#define BB 8
#define NN 2048
#define FF 2
#define DD 256
#define KSTEPS 10
#define OUTD 15

typedef _Float16 v16h __attribute__((ext_vector_type(16)));
typedef _Float16 v8h  __attribute__((ext_vector_type(8)));
typedef float    v8f  __attribute__((ext_vector_type(8)));

// LDS tile geometry: K-chunk of 64, row stride padded to 72 f16 (144B = 16B aligned,
// 36-dword row stride -> 16 consecutive rows hit 16 distinct banks)
#define KC   64
#define LSTR 72
#define SA_ELEMS (128 * LSTR)
#define SB_ELEMS (64 * LSTR)

static __device__ __forceinline__ v8f wmma_f16(v16h a, v16h b, v8f c) {
  // D = A(16x32 f16) * B(32x16 f16) + C(f32)
  return __builtin_amdgcn_wmma_f32_16x16x32_f16(false, a, false, b, (short)0, c, false, false);
}

// A-fragment (16x32 f16): lanes 0-15 hold M=lane, K=0..7 & 16..23; lanes 16-31 M=lane-16, K=8..15 & 24..31
static __device__ __forceinline__ v16h fragA(const _Float16* s, int Mo, int lr, int lh, int ss) {
  const _Float16* p = s + (Mo + lr) * LSTR + ss * 32 + 8 * lh;
  v8h lo = *(const v8h*)p;
  v8h hi = *(const v8h*)(p + 16);
  return __builtin_shufflevector(lo, hi, 0,1,2,3,4,5,6,7,8,9,10,11,12,13,14,15);
}

// B-fragment (32x16 f16) from K-major tile: lane holds col N=lane%16, K = 8*lh + e (e<8), 16 + 8*lh + (e-8) (e>=8)
static __device__ __forceinline__ v16h fragB(const _Float16* s, int No, int lr, int lh, int ss) {
  const _Float16* p = s + (No + lr) * LSTR + ss * 32 + 8 * lh;
  v8h lo = *(const v8h*)p;
  v8h hi = *(const v8h*)(p + 16);
  return __builtin_shufflevector(lo, hi, 0,1,2,3,4,5,6,7,8,9,10,11,12,13,14,15);
}

static __device__ __forceinline__ void load_regs(const _Float16* gAp, const _Float16* gBp,
                                                 int chunk, uint4 qa[4], uint4 qb[2]) {
  const uint4* pa = (const uint4*)(gAp + (size_t)chunk * KC);
  qa[0] = pa[0]; qa[1] = pa[1]; qa[2] = pa[2]; qa[3] = pa[3];
  const uint4* pb = (const uint4*)(gBp + (size_t)chunk * KC);
  qb[0] = pb[0]; qb[1] = pb[1];
}

static __device__ __forceinline__ void lds_fill(_Float16* sAbuf, _Float16* sBbuf,
                                                const uint4 qa[4], const uint4 qb[2],
                                                int ar, int acb, int br, int bcb) {
  uint4* da = (uint4*)&sAbuf[ar * LSTR + acb];
  da[0] = qa[0]; da[1] = qa[1]; da[2] = qa[2]; da[3] = qa[3];
  uint4* db = (uint4*)&sBbuf[br * LSTR + bcb];
  db[0] = qb[0]; db[1] = qb[1];
}

static __device__ __forceinline__ void compute_chunk(const _Float16* sAbuf, const _Float16* sBbuf,
                                                     int wm, int wn, int lr, int lh, v8f acc[2][2]) {
#pragma unroll
  for (int ss = 0; ss < 2; ++ss) {
    v16h a0 = fragA(sAbuf, wm * 32,      lr, lh, ss);
    v16h a1 = fragA(sAbuf, wm * 32 + 16, lr, lh, ss);
    v16h b0 = fragB(sBbuf, wn * 32,      lr, lh, ss);
    v16h b1 = fragB(sBbuf, wn * 32 + 16, lr, lh, ss);
    acc[0][0] = wmma_f16(a0, b0, acc[0][0]);
    acc[0][1] = wmma_f16(a0, b1, acc[0][1]);
    acc[1][0] = wmma_f16(a1, b0, acc[1][0]);
    acc[1][1] = wmma_f16(a1, b1, acc[1][1]);
  }
}

// Double-buffered, software-pipelined 128x64 workgroup-tile GEMM core.
// Buffer selection is COMPILE-TIME (loop unrolled by 2) so all LDS accesses stay in
// addrspace(3) -> ds_load_b128 / ds_store_b128 (no flat fallback).
// ktot must be a multiple of 128 (even number of 64-K chunks): 2048 and 256 both qualify.
static __device__ __forceinline__ void gemm_core(const _Float16* __restrict__ gA, size_t strideA,
                                                 const _Float16* __restrict__ gB, size_t strideB,
                                                 int ktot, _Float16* smem,
                                                 int tid, int wm, int wn, int lr, int lh,
                                                 v8f acc[2][2]) {
  _Float16* sA0 = smem;
  _Float16* sA1 = smem + SA_ELEMS;
  _Float16* sB0 = smem + 2 * SA_ELEMS;
  _Float16* sB1 = smem + 2 * SA_ELEMS + SB_ELEMS;

  const int ar = tid >> 1, acb = (tid & 1) * 32;   // A loader: 128 rows x 64 cols, 64B/thread
  const int br = tid >> 2, bcb = (tid & 3) * 16;   // B loader: 64 rows x 64 cols, 32B/thread
  const _Float16* gAp = gA + (size_t)ar * strideA + acb;
  const _Float16* gBp = gB + (size_t)br * strideB + bcb;

  uint4 qa[4], qb[2];
  const int nch = ktot / KC;   // even

  load_regs(gAp, gBp, 0, qa, qb);
  lds_fill(sA0, sB0, qa, qb, ar, acb, br, bcb);
  __syncthreads();

#pragma unroll 1
  for (int c = 0; c < nch; c += 2) {
    // ---- even chunk: compute buf0, fill buf1 with chunk c+1 ----
    load_regs(gAp, gBp, c + 1, qa, qb);
    if (c + 2 < nch) {  // warm WGP$ ahead (global_prefetch_b8)
      __builtin_prefetch(gAp + (size_t)(c + 2) * KC, 0, 1);
      __builtin_prefetch(gBp + (size_t)(c + 2) * KC, 0, 1);
    }
    compute_chunk(sA0, sB0, wm, wn, lr, lh, acc);
    lds_fill(sA1, sB1, qa, qb, ar, acb, br, bcb);
    __syncthreads();

    // ---- odd chunk: compute buf1, fill buf0 with chunk c+2 (if any) ----
    const bool more = (c + 2 < nch);
    if (more) {
      load_regs(gAp, gBp, c + 2, qa, qb);
      if (c + 3 < nch) {
        __builtin_prefetch(gAp + (size_t)(c + 3) * KC, 0, 1);
        __builtin_prefetch(gBp + (size_t)(c + 3) * KC, 0, 1);
      }
    }
    compute_chunk(sA1, sB1, wm, wn, lr, lh, acc);
    if (more) lds_fill(sA0, sB0, qa, qb, ar, acb, br, bcb);
    __syncthreads();
  }
}

// ---------------- 1) per-node embedding: H = tanh(feat @ W_emb[role] + b_emb[role]) ----------------
__global__ __launch_bounds__(256) void k_embed(const int* __restrict__ bus,
                                               const float* __restrict__ feat,
                                               const float* __restrict__ W,
                                               const float* __restrict__ be,
                                               float* __restrict__ Hf,
                                               _Float16* __restrict__ HT) {
  size_t idx = (size_t)blockIdx.x * 256 + threadIdx.x;   // over B*N*D
  int d = (int)(idx % DD);
  size_t bn = idx / DD;
  int n = (int)(bn % NN);
  int b = (int)(bn / NN);
  int r = bus[bn] - 1;
  float f0 = feat[bn * FF + 0], f1 = feat[bn * FF + 1];
  float w0 = W[((size_t)r * FF + 0) * DD + d];
  float w1 = W[((size_t)r * FF + 1) * DD + d];
  float h = tanhf(f0 * w0 + f1 * w1 + be[(size_t)r * DD + d]);
  Hf[idx] = h;
  HT[((size_t)b * DD + d) * NN + n] = (_Float16)h;
}

// ---------------- 2) adjacency: A = rownorm(mask(Yr|Yi), diag=0), stored f16 ----------------
__global__ __launch_bounds__(256) void k_adj(const float* __restrict__ Yr,
                                             const float* __restrict__ Yi,
                                             _Float16* __restrict__ Aout) {
  __shared__ int scnt;
  const int tid = threadIdx.x;
  const int row = blockIdx.x % NN;
  const int b   = blockIdx.x / NN;
  const size_t base = ((size_t)b * NN + row) * NN;
  if (tid == 0) scnt = 0;
  __syncthreads();
  bool msk[NN / 256];
  int cnt = 0;
#pragma unroll
  for (int i = 0; i < NN / 256; ++i) {
    int c = i * 256 + tid;
    bool m = ((Yr[base + c] != 0.0f) || (Yi[base + c] != 0.0f)) && (c != row);
    msk[i] = m;
    cnt += m ? 1 : 0;
  }
  atomicAdd(&scnt, cnt);
  __syncthreads();
  float inv = 1.0f / fmaxf((float)scnt, 1.0f);
#pragma unroll
  for (int i = 0; i < NN / 256; ++i) {
    int c = i * 256 + tid;
    Aout[base + c] = (_Float16)(msk[i] ? inv : 0.0f);
  }
}

// ---------------- 3) Wl transpose to f16 (B-operand wants K-major rows) ----------------
__global__ __launch_bounds__(256) void k_wlt(const float* __restrict__ Wl, _Float16* __restrict__ WlT) {
  int idx = blockIdx.x * 256 + threadIdx.x;   // over D*D
  int din = idx >> 8, dout = idx & 255;
  WlT[(size_t)dout * DD + din] = (_Float16)Wl[(size_t)din * DD + dout];
}

// ---------------- 4) AH = A @ H  (f16 WMMA, f32 accum), per-batch 2048x256, K=2048 ----------------
__global__ __launch_bounds__(256) void k_spmm(const _Float16* __restrict__ A,
                                              const _Float16* __restrict__ HT,
                                              _Float16* __restrict__ AH) {
  __shared__ __attribute__((aligned(16))) _Float16 smem[2 * SA_ELEMS + 2 * SB_ELEMS];
  const int tid = threadIdx.x;
  const int b = blockIdx.z;
  const int rowBase = blockIdx.x * 128;
  const int dBase   = blockIdx.y * 64;
  const int lane = tid & 31, wave = tid >> 5;
  const int wm = wave & 3, wn = wave >> 2;    // 4x2 waves -> 128x64 WG tile
  const int lh = lane >> 4, lr = lane & 15;

  v8f acc[2][2] = {};
  const _Float16* gA = A  + ((size_t)b * NN + rowBase) * NN;
  const _Float16* gB = HT + ((size_t)b * DD + dBase) * NN;

  gemm_core(gA, NN, gB, NN, NN, smem, tid, wm, wn, lr, lh, acc);

  // store f16 AH[b, row, col]; D-layout: M = j + 8*lh, N = lr
#pragma unroll
  for (int mi = 0; mi < 2; ++mi)
#pragma unroll
    for (int ni = 0; ni < 2; ++ni) {
      int col = dBase + wn * 32 + ni * 16 + lr;
      int rb  = rowBase + wm * 32 + mi * 16 + 8 * lh;
      _Float16* o = AH + ((size_t)b * NN + rb) * DD + col;
#pragma unroll
      for (int j = 0; j < 8; ++j) o[(size_t)j * DD] = (_Float16)acc[mi][ni][j];
    }
}

// ---------------- 5) H = H + tanh(AH @ Wl + bl); also refresh HT (f16, transposed) ----------------
__global__ __launch_bounds__(256) void k_update(const _Float16* __restrict__ AH,
                                                const _Float16* __restrict__ WlT,
                                                const float* __restrict__ bl,
                                                float* __restrict__ Hf,
                                                _Float16* __restrict__ HT) {
  __shared__ __attribute__((aligned(16))) _Float16 smem[2 * SA_ELEMS + 2 * SB_ELEMS];
  const int tid = threadIdx.x;
  const int b = blockIdx.z;
  const int rowBase = blockIdx.x * 128;
  const int dBase   = blockIdx.y * 64;
  const int lane = tid & 31, wave = tid >> 5;
  const int wm = wave & 3, wn = wave >> 2;
  const int lh = lane >> 4, lr = lane & 15;

  v8f acc[2][2] = {};
  const _Float16* gA = AH + ((size_t)b * NN + rowBase) * DD;
  const _Float16* gB = WlT + (size_t)dBase * DD;

  gemm_core(gA, DD, gB, DD, DD, smem, tid, wm, wn, lr, lh, acc);

#pragma unroll
  for (int mi = 0; mi < 2; ++mi)
#pragma unroll
    for (int ni = 0; ni < 2; ++ni) {
      int col = dBase + wn * 32 + ni * 16 + lr;
      int rb  = rowBase + wm * 32 + mi * 16 + 8 * lh;
      float bias = bl[col];
#pragma unroll
      for (int j = 0; j < 8; ++j) {
        int row = rb + j;
        size_t idx = ((size_t)b * NN + row) * DD + col;
        float v = Hf[idx] + tanhf(acc[mi][ni][j] + bias);
        Hf[idx] = v;
        HT[((size_t)b * DD + col) * NN + row] = (_Float16)v;
      }
    }
}

// ---------------- 6) out[b] = mean_n H[b,n,:] @ Wd + bd ----------------
__global__ __launch_bounds__(256) void k_decode(const float* __restrict__ Hf,
                                                const float* __restrict__ Wd,
                                                const float* __restrict__ bd,
                                                float* __restrict__ out) {
  __shared__ float sM[DD];
  const int b = blockIdx.x, d = threadIdx.x;
  float s = 0.0f;
  const float* p = Hf + (size_t)b * NN * DD + d;
  for (int m = 0; m < NN; ++m) s += p[(size_t)m * DD];
  sM[d] = s * (1.0f / (float)NN);
  __syncthreads();
  if (d < OUTD) {
    float a = bd[d];
    for (int k = 0; k < DD; ++k) a += sM[k] * Wd[(size_t)k * OUTD + d];
    out[(size_t)b * OUTD + d] = a;
  }
}

extern "C" void kernel_launch(void* const* d_in, const int* in_sizes, int n_in,
                              void* d_out, int out_size, void* d_ws, size_t ws_size,
                              hipStream_t stream) {
  (void)in_sizes; (void)n_in; (void)out_size; (void)ws_size;
  const int*   bus  = (const int*)d_in[0];
  const float* Yr   = (const float*)d_in[1];
  const float* Yi   = (const float*)d_in[2];
  const float* feat = (const float*)d_in[3];
  const float* Wemb = (const float*)d_in[4];
  const float* bemb = (const float*)d_in[5];
  const float* Wl   = (const float*)d_in[6];
  const float* bl   = (const float*)d_in[7];
  const float* Wd   = (const float*)d_in[8];
  const float* bd   = (const float*)d_in[9];
  float* out = (float*)d_out;

  char* ws = (char*)d_ws;
  size_t off = 0;
  auto take = [&](size_t bytes) -> char* {
    char* p = ws + off;
    off = (off + bytes + 255) & ~(size_t)255;
    return p;
  };
  float*    Hf  = (float*)   take((size_t)BB * NN * DD * sizeof(float));     // 16.8 MB
  _Float16* HT  = (_Float16*)take((size_t)BB * DD * NN * sizeof(_Float16));  // 8.4 MB
  _Float16* Ah  = (_Float16*)take((size_t)BB * NN * DD * sizeof(_Float16));  // 8.4 MB
  _Float16* Am  = (_Float16*)take((size_t)BB * NN * NN * sizeof(_Float16));  // 67 MB (L2-resident)
  _Float16* WlT = (_Float16*)take((size_t)DD * DD * sizeof(_Float16));

  k_embed<<<dim3((BB * NN * DD) / 256), 256, 0, stream>>>(bus, feat, Wemb, bemb, Hf, HT);
  k_adj  <<<dim3(BB * NN),              256, 0, stream>>>(Yr, Yi, Am);
  k_wlt  <<<dim3((DD * DD) / 256),      256, 0, stream>>>(Wl, WlT);

  dim3 g(NN / 128, DD / 64, BB);   // 16 x 4 x 8
  for (int s = 0; s < KSTEPS; ++s) {
    k_spmm  <<<g, 256, 0, stream>>>(Am, HT, Ah);
    k_update<<<g, 256, 0, stream>>>(Ah, WlT, bl, Hf, HT);
  }
  k_decode<<<dim3(BB), 256, 0, stream>>>(Hf, Wd, bd, out);
}
